// DIEN_4535485464623
// MI455X (gfx1250) — matrix operand order repaired and checked by reading, not compiled, not run on hardware
//
#include <hip/hip_runtime.h>
#include <cstdint>

#define BB 1024
#define TT 200
#define HH 128

typedef __attribute__((ext_vector_type(16))) __bf16 v16bf;
typedef __attribute__((ext_vector_type(8)))  float  v8f;
typedef __attribute__((ext_vector_type(4)))  unsigned u32x4;
typedef __attribute__((ext_vector_type(8)))  int      i32x8;
typedef __attribute__((ext_vector_type(4)))  int      i32x4;

static __device__ __forceinline__ v8f wmma_bf16(v16bf a, v16bf b, v8f c) {
  // D = A(16x32 bf16) x B(32x16 bf16) + C(16x16 f32)
  return __builtin_amdgcn_wmma_f32_16x16x32_bf16(false, a, false, b, (short)0, c,
                                                 false, false);
}

// Pack two f32 -> two bf16 in one 32-bit word (lowers to v_cvt_pk_bf16_f32).
static __device__ __forceinline__ unsigned pack_bf2(float lo, float hi) {
  unsigned short l = __builtin_bit_cast(unsigned short, (__bf16)lo);
  unsigned short h = __builtin_bit_cast(unsigned short, (__bf16)hi);
  return (unsigned)l | ((unsigned)h << 16);
}

// bf16 A-tile in LDS, pre-swizzled to WMMA A-fragment layout (ISA 7.12.2):
// tile[kk][slot(32)][16 bf16], slot = m + 16*kh. Element address for (m, k):
static __device__ __forceinline__ int a_elem_addr(int m, int k) {
  int kk = k >> 5, kp = k & 31;
  int kh = (kp >> 3) & 1;
  int e  = (kp & 7) | ((kp >> 4) << 3);
  return (kk * 32 + m + 16 * kh) * 16 + e;
}

// A-fragment load: lane's 16 bf16 are 32 contiguous bytes -> 2x ds_load_b128, 0 VALU.
static __device__ __forceinline__ v16bf load_A_frag(const __bf16* t, int lane, int kk) {
  return *(const v16bf*)(t + (kk * 32 + lane) * 16);
}

// B fragment (32x16 bf16) from global f32 weight W[N][K] (B[k][n] = W[n][k]).
static __device__ __forceinline__ v16bf load_B(const float* W, int ldk, int lane,
                                               int c0, int kk) {
  int n = c0 + (lane & 15), kh = lane >> 4;
  const float* row = W + (size_t)n * ldk + kk * 32 + 8 * kh;
  v16bf b;
#pragma unroll
  for (int r = 0; r < 8; ++r) {
    b[r]     = (__bf16)row[r];
    b[r + 8] = (__bf16)row[16 + r];
  }
  return b;
}

static __device__ __forceinline__ float sigm(float x) {
  return 1.0f / (1.0f + __expf(-x));
}

// ---- TDM: DMA a 16-row x 128-col f32 tile (row stride in elems) into LDS ----
// D# per ISA 8.3/8.4: group0 = {flags, lds_addr, global_addr lo/hi|type=2},
// group1 = {mask/data_size, dims/tiles, strides}. 2-D tensor -> groups 2/3 zero.
static __device__ __forceinline__ void tdm_load_tile16x128(unsigned lds_off,
                                                           const float* gptr,
                                                           int rowStrideElems) {
  unsigned long long ga = (unsigned long long)(uintptr_t)gptr;
  u32x4 g0;
  g0[0] = 1u;                                          // count=1 (valid), user D#
  g0[1] = lds_off;                                     // lds_addr (bytes)
  g0[2] = (unsigned)(ga & 0xffffffffu);                // global_addr[31:0]
  g0[3] = (unsigned)((ga >> 32) & 0x01ffffffu)         // global_addr[56:32]
          | (2u << 30);                                // type = 2 ("image")
  i32x8 g1;
  g1[0] = 0x00020000;          // workgroup_mask=0 | data_size=2 (4 bytes)
  g1[1] = (int)(128u << 16);   // atomic_barrier=0 | tensor_dim0[15:0]=128
  g1[2] = (int)(16u << 16);    // tensor_dim0[31:16]=0 | tensor_dim1[15:0]=16
  g1[3] = (int)(128u << 16);   // tensor_dim1[31:16]=0 | tile_dim0=128
  g1[4] = 16;                  // tile_dim1=16 | tile_dim2=0
  g1[5] = rowStrideElems;      // tensor_dim0_stride[31:0]
  g1[6] = 0;                   // stride hi | tensor_dim1_stride lo
  g1[7] = 0;
  i32x4 gz = {0, 0, 0, 0};
#if __clang_major__ >= 23
  i32x8 gz8 = {0, 0, 0, 0, 0, 0, 0, 0};
  __builtin_amdgcn_tensor_load_to_lds(g0, g1, gz, gz, gz8, 0);
#else
  __builtin_amdgcn_tensor_load_to_lds(g0, g1, gz, gz, 0);
#endif
}

// ---------------- Kernel 1: attention logits ----------------
// logits[bt] = sum_h (targets[bt,:]·W_w[h,:] + W_b[h]) * history[bt,h]
__global__ __launch_bounds__(256) void attn_logits(const float* __restrict__ tgt,
                                                   const float* __restrict__ hist,
                                                   const float* __restrict__ Ww,
                                                   const float* __restrict__ Wb,
                                                   float* __restrict__ logits) {
  __shared__ __bf16 tgA[4 * 32 * 16];   // targets tile, bf16 A-layout
  __shared__ float  hi[16 * 132];       // history tile, f32 (elementwise use)
  __shared__ float  red[16];
  int tid = threadIdx.x;
  int row0 = blockIdx.x * 16;
  for (int idx = tid; idx < 16 * 64; idx += 256) {      // (m, k-pair)
    int m = idx >> 6, k = (idx & 63) * 2;
    const float* s = &tgt[(size_t)(row0 + m) * 128 + k];
    *(unsigned*)&tgA[a_elem_addr(m, k)] = pack_bf2(s[0], s[1]);
  }
  for (int idx = tid; idx < 16 * 128; idx += 256) {
    int m = idx >> 7, k = idx & 127;
    hi[m * 132 + k] = hist[(size_t)(row0 + m) * 128 + k];
  }
  if (tid < 16) red[tid] = 0.0f;
  __syncthreads();

  int lane = tid & 31, wid = tid >> 5, c0 = wid * 16;
  v8f acc = {0.f, 0.f, 0.f, 0.f, 0.f, 0.f, 0.f, 0.f};
#pragma unroll
  for (int kk = 0; kk < 4; ++kk) {
    v16bf a = load_A_frag(tgA, lane, kk);
    v16bf b = load_B(Ww, 128, lane, c0, kk);
    acc = wmma_bf16(a, b, acc);
  }
  int n = c0 + (lane & 15), mh = lane >> 4;
  float wb = Wb[n];
#pragma unroll
  for (int r = 0; r < 8; ++r) {
    int m = r + 8 * mh;
    float v = (acc[r] + wb) * hi[m * 132 + n];
    v += __shfl_xor(v, 1);
    v += __shfl_xor(v, 2);
    v += __shfl_xor(v, 4);
    v += __shfl_xor(v, 8);
    if ((lane & 15) == 0) atomicAdd(&red[m], v);  // ds_add_f32
  }
  __syncthreads();
  if (tid < 16) logits[row0 + tid] = red[tid];
}

// ---------------- Kernel 2: softmax over T (in place) ----------------
__global__ __launch_bounds__(256) void softmax_t(float* __restrict__ att) {
  __shared__ float buf[256];
  int b = blockIdx.x, tid = threadIdx.x;
  float x = (tid < TT) ? att[(size_t)b * TT + tid] : -3.0e38f;
  buf[tid] = x;
  __syncthreads();
  for (int s = 128; s > 0; s >>= 1) {
    if (tid < s) buf[tid] = fmaxf(buf[tid], buf[tid + s]);
    __syncthreads();
  }
  float mx = buf[0];
  __syncthreads();
  float e = (tid < TT) ? __expf(x - mx) : 0.0f;
  buf[tid] = e;
  __syncthreads();
  for (int s = 128; s > 0; s >>= 1) {
    if (tid < s) buf[tid] += buf[tid + s];
    __syncthreads();
  }
  float inv = 1.0f / buf[0];
  if (tid < TT) att[(size_t)b * TT + tid] = e * inv;
}

// ---------------- Kernel 3: fused GRU scan with TDM prefetch ----------------
// Block owns 16 batch rows for all 200 steps. h lives in registers; bf16
// A-layout shadow in LDS feeds WMMA. Wave 0 DMAs x_{t+1} via the Tensor Data
// Mover while step t computes; conversion happens after s_wait_tensorcnt.
__global__ __launch_bounds__(256) void gru_scan(
    const float* __restrict__ hist, const float* __restrict__ att,
    const float* __restrict__ xu_w, const float* __restrict__ xu_b,
    const float* __restrict__ hu_w, const float* __restrict__ hu_b,
    const float* __restrict__ xr_w, const float* __restrict__ xr_b,
    const float* __restrict__ hr_w, const float* __restrict__ hr_b,
    const float* __restrict__ xg_w, const float* __restrict__ xg_b,
    const float* __restrict__ hg_w, const float* __restrict__ hg_b,
    float* __restrict__ hfin) {
  __shared__ __bf16 hA[4 * 32 * 16];    // h, bf16 A-layout
  __shared__ __bf16 xA[4 * 32 * 16];    // x_t, bf16 A-layout
  __shared__ float  xstage[16 * 128];   // TDM landing buffer (f32, row-major)
  __shared__ float  att_s[TT * 16];     // att[b0+m][t]

  int tid = threadIdx.x;
  int b0 = blockIdx.x * 16;
  for (int idx = tid; idx < 1024; idx += 256) ((unsigned*)hA)[idx] = 0u;  // h0 = 0
  for (int idx = tid; idx < 16 * TT; idx += 256) {
    int m = idx / TT, t = idx - m * TT;
    att_s[t * 16 + m] = att[(size_t)(b0 + m) * TT + t];
  }
  int lane = tid & 31, wid = tid >> 5, c0 = wid * 16;
  int n = c0 + (lane & 15), mh = lane >> 4;
  unsigned xstage_off = (unsigned)(uintptr_t)&xstage[0];  // LDS byte offset

  // Register-resident bf16 B-fragments for all six 128x128 weights.
  v16bf Bhu[4], Bxu[4], Bhr[4], Bxr[4], Bhg[4], Bxg[4];
#pragma unroll
  for (int kk = 0; kk < 4; ++kk) {
    Bhu[kk] = load_B(hu_w, 128, lane, c0, kk);
    Bxu[kk] = load_B(xu_w, 128, lane, c0, kk);
    Bhr[kk] = load_B(hr_w, 128, lane, c0, kk);
    Bxr[kk] = load_B(xr_w, 128, lane, c0, kk);
    Bhg[kk] = load_B(hg_w, 128, lane, c0, kk);
    Bxg[kk] = load_B(xg_w, 128, lane, c0, kk);
  }
  float bu = xu_b[n] + hu_b[n];
  float br = xr_b[n] + hr_b[n];
  float bs = hg_b[n];
  float bx = xg_b[n];

  int hbase = a_elem_addr(0, n);  // per-lane scatter base for h[m][n] (k == n)
  v8f hreg = {0.f, 0.f, 0.f, 0.f, 0.f, 0.f, 0.f, 0.f};

  // Prologue: DMA x_0 and convert it into bf16 A-fragments.
  if (wid == 0) {
    tdm_load_tile16x128(xstage_off, &hist[(size_t)b0 * TT * HH], TT * HH);
    __builtin_amdgcn_s_wait_tensorcnt(0);
  }
  __syncthreads();
  for (int idx = tid; idx < 16 * 64; idx += 256) {
    int m = idx >> 6, k = (idx & 63) * 2;
    const float* s = &xstage[m * 128 + k];
    *(unsigned*)&xA[a_elem_addr(m, k)] = pack_bf2(s[0], s[1]);
  }

  for (int t = 0; t < TT; ++t) {
    __syncthreads();  // B1: xA/hA fragments of step t visible to all waves
    if (wid == 0 && t + 1 < TT)   // kick DMA for x_{t+1}; overlaps this step
      tdm_load_tile16x128(xstage_off, &hist[((size_t)b0 * TT + (t + 1)) * HH],
                          TT * HH);

    v8f aU = {0.f, 0.f, 0.f, 0.f, 0.f, 0.f, 0.f, 0.f};
    v8f aR = aU, aS = aU, aX = aU;
#pragma unroll
    for (int kk = 0; kk < 4; ++kk) {
      v16bf ah = load_A_frag(hA, lane, kk);
      v16bf ax = load_A_frag(xA, lane, kk);
      aU = wmma_bf16(ah, Bhu[kk], aU);
      aU = wmma_bf16(ax, Bxu[kk], aU);
      aR = wmma_bf16(ah, Bhr[kk], aR);
      aR = wmma_bf16(ax, Bxr[kk], aR);
      aS = wmma_bf16(ah, Bhg[kk], aS);   // h@hg^T
      aX = wmma_bf16(ax, Bxg[kk], aX);   // x@xg^T
    }
#pragma unroll
    for (int r = 0; r < 8; ++r) {
      int m = r + 8 * mh;
      float a  = att_s[t * 16 + m];
      float u  = sigm(aU[r] + bu) * a;
      float rr = sigm(aR[r] + br);
      float g  = tanhf(aX[r] + bx + rr * (aS[r] + bs));
      float ho = hreg[r];
      hreg[r] = ho + u * (g - ho);
    }
    __syncthreads();  // B2: all waves finished reading hA/xA fragments
#pragma unroll
    for (int r = 0; r < 8; ++r)
      hA[hbase + (r + 8 * mh) * 16] = (__bf16)hreg[r];

    if (t + 1 < TT) {
      if (wid == 0) __builtin_amdgcn_s_wait_tensorcnt(0);  // DMA landed
      __syncthreads();  // B3: xstage contents visible to every wave
      for (int idx = tid; idx < 16 * 64; idx += 256) {     // convert for t+1
        int m = idx >> 6, k = (idx & 63) * 2;
        const float* s = &xstage[m * 128 + k];
        *(unsigned*)&xA[a_elem_addr(m, k)] = pack_bf2(s[0], s[1]);
      }
    }
  }
  // h_final straight from registers
#pragma unroll
  for (int r = 0; r < 8; ++r)
    hfin[(size_t)(b0 + r + 8 * mh) * 128 + n] = hreg[r];
}

// ---------------- Kernel 4: out = concat(h, targets[:,0]) @ ln2_w^T + b ----------------
__global__ __launch_bounds__(256) void out_gemm(const float* __restrict__ hfin,
                                                const float* __restrict__ tgt,
                                                const float* __restrict__ Lw,
                                                const float* __restrict__ Lb,
                                                float* __restrict__ out) {
  __shared__ __bf16 aA[8 * 32 * 16];   // K = 256, bf16 A-layout
  int tid = threadIdx.x;
  int b0 = blockIdx.x * 16;
  for (int idx = tid; idx < 16 * 128; idx += 256) {     // (m, k-pair)
    int m = idx >> 7, k = (idx & 127) * 2;
    float lo, hi;
    if (k < 128) {
      const float* s = &hfin[(size_t)(b0 + m) * 128 + k];
      lo = s[0]; hi = s[1];
    } else {
      const float* s = &tgt[(size_t)(b0 + m) * TT * 128 + (k - 128)];
      lo = s[0]; hi = s[1];
    }
    *(unsigned*)&aA[a_elem_addr(m, k)] = pack_bf2(lo, hi);
  }
  __syncthreads();
  int lane = tid & 31, wid = tid >> 5, c0 = wid * 16;
  v8f acc = {0.f, 0.f, 0.f, 0.f, 0.f, 0.f, 0.f, 0.f};
#pragma unroll
  for (int kk = 0; kk < 8; ++kk) {
    v16bf a = load_A_frag(aA, lane, kk);
    v16bf b = load_B(Lw, 256, lane, c0, kk);
    acc = wmma_bf16(a, b, acc);
  }
  int n = c0 + (lane & 15), mh = lane >> 4;
  float lb = Lb[n];
#pragma unroll
  for (int r = 0; r < 8; ++r)
    out[(size_t)(b0 + r + 8 * mh) * 128 + n] = acc[r] + lb;
}

extern "C" void kernel_launch(void* const* d_in, const int* in_sizes, int n_in,
                              void* d_out, int out_size, void* d_ws, size_t ws_size,
                              hipStream_t stream) {
  const float* tgt  = (const float*)d_in[0];
  const float* hist = (const float*)d_in[1];
  const float* Ww   = (const float*)d_in[2];
  const float* Wb   = (const float*)d_in[3];
  const float* xu_w = (const float*)d_in[4];
  const float* xu_b = (const float*)d_in[5];
  const float* hu_w = (const float*)d_in[6];
  const float* hu_b = (const float*)d_in[7];
  const float* xr_w = (const float*)d_in[8];
  const float* xr_b = (const float*)d_in[9];
  const float* hr_w = (const float*)d_in[10];
  const float* hr_b = (const float*)d_in[11];
  const float* xg_w = (const float*)d_in[12];
  const float* xg_b = (const float*)d_in[13];
  const float* hg_w = (const float*)d_in[14];
  const float* hg_b = (const float*)d_in[15];
  const float* Lw   = (const float*)d_in[16];
  const float* Lb   = (const float*)d_in[17];

  float* ws   = (float*)d_ws;
  float* att  = ws;                         // B*T f32 (logits -> softmax in place)
  float* hfin = ws + (size_t)BB * TT;       // B*H f32

  attn_logits<<<(BB * TT) / 16, 256, 0, stream>>>(tgt, hist, Ww, Wb, att);
  softmax_t<<<BB, 256, 0, stream>>>(att);
  gru_scan<<<BB / 16, 256, 0, stream>>>(hist, att, xu_w, xu_b, hu_w, hu_b,
                                        xr_w, xr_b, hr_w, hr_b,
                                        xg_w, xg_b, hg_w, hg_b, hfin);
  out_gemm<<<BB / 16, 256, 0, stream>>>(hfin, tgt, Lw, Lb, (float*)d_out);
}
